// JustAttention2GCN_50130858279704
// MI455X (gfx1250) — matrix operand order repaired and checked by reading, not compiled, not run on hardware
//
#include <hip/hip_runtime.h>

typedef __bf16 bf16;
typedef __attribute__((ext_vector_type(16))) __bf16 v16bf;
typedef __attribute__((ext_vector_type(8)))  float  v8f;

// Model dims (from reference)
constexpr int Bb_    = 8;
constexpr int T_     = 20;
constexpr int N_     = 256;
constexpr int BN_    = 2048;      // B*N
constexpr int H_     = 128;
constexpr int HEADS_ = 8;
constexpr int DH_    = 16;
constexpr int FF_    = 512;
constexpr int LAYERS_= 5;
constexpr int ROWS_  = T_ * BN_;  // 40960 flattened (t, node) rows

// ---------------------------------------------------------------- wave utils
__device__ __forceinline__ float wave_sum(float v) {
#pragma unroll
  for (int o = 16; o > 0; o >>= 1) v += __shfl_xor(v, o, 32);
  return v;
}
__device__ __forceinline__ float wave_max(float v) {
#pragma unroll
  for (int o = 16; o > 0; o >>= 1) v = fmaxf(v, __shfl_xor(v, o, 32));
  return v;
}

union FragU { v16bf v; unsigned int u[8]; };
union Pack4 { uint2 q; bf16 h[4]; };

// ---------------------------------------------------------------- small prep kernels
__global__ void mask_kernel(const unsigned char* __restrict__ ego, float* __restrict__ maskf) {
  int idx = blockIdx.x * 256 + threadIdx.x;
  if (idx >= T_ * BN_) return;
  int t = idx / BN_, bn = idx % BN_;
  int b = bn / N_, n = bn % N_;
  maskf[idx] = ego[(long long)b * T_ * N_ + (long long)t * N_ + n] ? 1.f : 0.f;
}

__global__ void f2bf_kernel(const float* __restrict__ in, bf16* __restrict__ out, long long n) {
  long long i = (long long)blockIdx.x * 256 + threadIdx.x;
  if (i < n) out[i] = (bf16)in[i];
}

// per-matrix transpose convert: out[m][n][k] = in[m][k][n]   (H_ x H_ matrices)
__global__ void f2bf_t_kernel(const float* __restrict__ in, bf16* __restrict__ out) {
  int m = blockIdx.y;
  int idx = blockIdx.x * 256 + threadIdx.x;
  if (idx >= H_ * H_) return;
  int n = idx >> 7, k = idx & (H_ - 1);
  out[(long long)m * H_ * H_ + idx] = (bf16)in[(long long)m * H_ * H_ + (long long)k * H_ + n];
}

// deg[j] = m_j * (sum_i A[i][j]*m_i + 1);  dinv = deg>0 ? rsqrt(deg) : 0
__global__ __launch_bounds__(256) void deg_dinv_kernel(const float* __restrict__ adj,
                                                       const float* __restrict__ maskf,
                                                       float* __restrict__ dinv) {
  __shared__ float sm[BN_];
  const int t = blockIdx.y;
  for (int i = threadIdx.x; i < BN_; i += 256) sm[i] = maskf[t * BN_ + i];
  __syncthreads();
  const int j = blockIdx.x * 256 + threadIdx.x;
  const float* At = adj + (long long)t * BN_ * BN_;
  float s = 0.f;
  for (int i = 0; i < BN_; ++i) s += At[(long long)i * BN_ + j] * sm[i];
  float mj = maskf[t * BN_ + j];
  float deg = mj * (s + 1.0f);
  dinv[t * BN_ + j] = deg > 0.f ? rsqrtf(deg) : 0.f;
}

// S[t][i][j] = dinv[i]*dinv[j]*(adj[t][j][i] + (i==j))  -- row-major S via LDS tile transpose
__global__ __launch_bounds__(256) void build_S_kernel(const float* __restrict__ adj,
                                                      const float* __restrict__ dinv,
                                                      bf16* __restrict__ S) {
  __shared__ float tile[64][68];     // 68-pad: float4-aligned rows, odd dword banking
  const int t  = blockIdx.z;
  const int i0 = blockIdx.y * 64;    // output row block
  const int j0 = blockIdx.x * 64;    // output col block
  const long long abase = (long long)t * BN_ * BN_;
  // read adj[t][j0+r][i0+c] coalesced (c fastest), float4 per thread
#pragma unroll
  for (int it = 0; it < 4; ++it) {
    int idx = threadIdx.x + it * 256;       // 1024 float4 slots
    int c4 = idx & 15, r = idx >> 4;
    *(float4*)&tile[r][c4 * 4] =
        *(const float4*)&adj[abase + (long long)(j0 + r) * BN_ + (i0 + c4 * 4)];
  }
  __syncthreads();
  // write S[t][i0+r][j0+c] coalesced, 4 bf16 per thread (uint2 store)
#pragma unroll
  for (int it = 0; it < 4; ++it) {
    int idx = threadIdx.x + it * 256;
    int c4 = idx & 15, r = idx >> 4;
    const int i = i0 + r;
    const float di = dinv[t * BN_ + i];
    Pack4 p;
#pragma unroll
    for (int e = 0; e < 4; ++e) {
      int j = j0 + c4 * 4 + e;
      float v = di * dinv[t * BN_ + j] * (tile[c4 * 4 + e][r] + (i == j ? 1.f : 0.f));
      p.h[e] = (bf16)v;
    }
    *(uint2*)&S[abase + (long long)i * BN_ + (j0 + c4 * 4)] = p.q;
  }
}

// Pt[t][h][node] = x_raw[t][node][:] @ gcn1_w   (K=2: plain VALU, transposed bf16 out)
__global__ void gcn_in_kernel(const float* __restrict__ xr, const float* __restrict__ W,
                              bf16* __restrict__ Pt) {
  long long idx = (long long)blockIdx.x * 256 + threadIdx.x;
  if (idx >= (long long)ROWS_ * H_) return;
  int node = (int)(idx % BN_);
  long long rest = idx / BN_;
  int h = (int)(rest % H_);
  int t = (int)(rest / H_);
  long long r = (long long)t * BN_ + node;
  Pt[idx] = (bf16)(xr[r * 2] * W[h] + xr[r * 2 + 1] * W[H_ + h]);
}

// ---------------------------------------------------------------- WMMA GEMM
// C[M,N] (opt +bias, opt relu) = A[M,K] @ B^T, A row-major bf16, B stored [N][K] bf16.
// Block: 256 thr = 8 waves (4 M x 2 N), block tile 64x64, BK=32.
// Each wave: one A frag, two B frags, two f32 accumulators -> 2 WMMA / phase.
// OUT_TRANS: store bf16 transposed per-timestep: Out[t][col][node] (GCN Pt geometry).
template <bool RELU, bool OUT_BF16, bool OUT_TRANS>
__global__ __launch_bounds__(256) void wmma_gemm_kernel(
    const bf16* __restrict__ A, const bf16* __restrict__ Bm,
    const float* __restrict__ bias, void* __restrict__ Out,
    int M, int Nn, int K, int lda, int ldb, int ldo,
    long long aBatch, long long bBatch, long long oBatch) {
  (void)M; (void)Nn;
  __shared__ bf16 sA[64][36];   // [m][k]  rows 72B -> b64-aligned, b32 pair reads
  __shared__ bf16 sBt[64][36];  // [n][k]  K-pairs contiguous
  const int tid  = threadIdx.x;
  const int lane = tid & 31;
  const int wave = tid >> 5;
  const int wm = wave >> 1;     // 0..3 -> M subtile
  const int wn = wave & 1;      // 0..1 -> 32-wide N slab
  const int m0 = blockIdx.x * 64;
  const int n0 = blockIdx.y * 64;
  const long long zb = blockIdx.z;
  const bf16* Ab = A  + zb * aBatch;
  const bf16* Bp = Bm + zb * bBatch;

  const int half = lane >> 4;   // K-half of the wave
  const int l16  = lane & 15;

  v8f acc0 = {0.f, 0.f, 0.f, 0.f, 0.f, 0.f, 0.f, 0.f};
  v8f acc1 = {0.f, 0.f, 0.f, 0.f, 0.f, 0.f, 0.f, 0.f};

  for (int k0 = 0; k0 < K; k0 += 32) {
    // ---- stage A (64x32) and B (64x32, [n][k]) tiles: all b64 traffic ----
#pragma unroll
    for (int it = 0; it < 2; ++it) {
      int idx = tid + it * 256;
      int kv = idx & 7, m = idx >> 3;          // 4 consecutive k per thread
      *(uint2*)&sA[m][kv * 4] =
          *(const uint2*)&Ab[(long long)(m0 + m) * lda + (k0 + kv * 4)];
      *(uint2*)&sBt[m][kv * 4] =
          *(const uint2*)&Bp[(long long)(n0 + m) * ldb + (k0 + kv * 4)];
    }
    // software prefetch of next K tile (global_prefetch_b8)
    if (k0 + 32 < K) {
      __builtin_prefetch(&Ab[(long long)(m0 + (tid >> 3)) * lda + (k0 + 32)], 0, 3);
      __builtin_prefetch(&Bp[(long long)(n0 + (tid >> 3)) * ldb + (k0 + 32)], 0, 3);
    }
    __syncthreads();

    // Fragments per documented CDNA5 wave32 layouts (b32 LDS reads):
    // A 16x32 bf16: lanes 0-15 -> K {0..7,16..23}; lanes 16-31 -> K {8..15,24..31}
    // B 32x16 bf16: lanes 0-15 -> K {0..7,16..23}; lanes 16-31 -> K {8..15,24..31}
    FragU af, bf0, bf1;
    const int arow = wm * 16 + l16;
    const int bcol = wn * 32 + l16;
#pragma unroll
    for (int v = 0; v < 8; ++v) {
      int kb = (v < 4) ? (half * 8 + 2 * v) : (16 + half * 8 + 2 * (v - 4));
      af.u[v] = *(const unsigned int*)&sA[arow][kb];
      int kk = kb;
      bf0.u[v] = *(const unsigned int*)&sBt[bcol][kk];
      bf1.u[v] = *(const unsigned int*)&sBt[bcol + 16][kk];
    }
    acc0 = __builtin_amdgcn_wmma_f32_16x16x32_bf16(false, af.v, false, bf0.v,
                                                   (short)0, acc0, false, false);
    acc1 = __builtin_amdgcn_wmma_f32_16x16x32_bf16(false, af.v, false, bf1.v,
                                                   (short)0, acc1, false, false);
    __syncthreads();
  }

  // C layout: VGPR r -> M = half*8 + r, N = lane&15
  const int col0 = n0 + wn * 32 + l16;
  const int col1 = col0 + 16;
  const float bv0 = bias ? bias[col0] : 0.f;
  const float bv1 = bias ? bias[col1] : 0.f;
#pragma unroll
  for (int r = 0; r < 8; ++r) {
    int row = m0 + wm * 16 + half * 8 + r;
    float v0 = acc0[r] + bv0;
    float v1 = acc1[r] + bv1;
    if (RELU) { v0 = fmaxf(v0, 0.f); v1 = fmaxf(v1, 0.f); }
    if (OUT_TRANS) {
      // Pt[t][col][node] for GCN geometry (M = T*BN rows of 2048)
      int tt = row >> 11, node = row & (BN_ - 1);
      long long tb = (long long)tt * H_ * BN_ + node;
      ((bf16*)Out)[tb + (long long)col0 * BN_] = (bf16)v0;
      ((bf16*)Out)[tb + (long long)col1 * BN_] = (bf16)v1;
    } else {
      long long off = zb * oBatch + (long long)row * ldo;
      if (OUT_BF16) {
        ((bf16*)Out)[off + col0] = (bf16)v0;
        ((bf16*)Out)[off + col1] = (bf16)v1;
      } else {
        ((float*)Out)[off + col0] = v0;
        ((float*)Out)[off + col1] = v1;
      }
    }
  }
}

// ---------------------------------------------------------------- LayerNorm (wave per row)
// y = LN(in + bias? + resPre?) * g + b ; then (+resPost?) ; then relu?
__global__ __launch_bounds__(256) void ln_kernel(
    const float* __restrict__ in, const float* __restrict__ bias,
    const float* __restrict__ resPre, const bf16* __restrict__ resPost,
    const float* __restrict__ g, const float* __restrict__ bta,
    float* __restrict__ outF, bf16* __restrict__ outB, int rows, int relu) {
  const int wave = threadIdx.x >> 5, lane = threadIdx.x & 31;
  const int row = blockIdx.x * 8 + wave;
  if (row >= rows) return;
  const long long base = (long long)row * H_;
  float v[4];
  float s = 0.f;
#pragma unroll
  for (int j = 0; j < 4; ++j) {
    int c = lane + 32 * j;
    float x = in[base + c];
    if (bias)   x += bias[c];
    if (resPre) x += resPre[base + c];
    v[j] = x;
    s += x;
  }
  s = wave_sum(s);
  const float mu = s * (1.f / H_);
  float vs = 0.f;
#pragma unroll
  for (int j = 0; j < 4; ++j) { float d = v[j] - mu; vs += d * d; }
  vs = wave_sum(vs);
  const float rstd = rsqrtf(vs * (1.f / H_) + 1e-5f);
#pragma unroll
  for (int j = 0; j < 4; ++j) {
    int c = lane + 32 * j;
    float y = (v[j] - mu) * rstd * g[c] + bta[c];
    if (resPost) y += (float)resPost[base + c];
    if (relu) y = fmaxf(y, 0.f);
    if (outF) outF[base + c] = y;
    if (outB) outB[base + c] = (bf16)y;
  }
}

// ---------------------------------------------------------------- attention (wave per node,head)
__global__ __launch_bounds__(32) void attn_kernel(const float* __restrict__ QKV,
                                                  const float* __restrict__ maskf,
                                                  bf16* __restrict__ o) {
  const int bn = blockIdx.x / HEADS_;
  const int h  = blockIdx.x % HEADS_;
  const int lane = threadIdx.x;

  float kv[DH_];
  float kb = -1e30f;
  if (lane < T_) {
    long long kr = ((long long)bn * T_ + lane) * (3 * H_) + H_ + h * DH_;
#pragma unroll
    for (int d = 0; d < DH_; ++d) kv[d] = QKV[kr + d];
    kb = maskf[lane * BN_ + bn] > 0.f ? 0.f : -1e30f;
  } else {
#pragma unroll
    for (int d = 0; d < DH_; ++d) kv[d] = 0.f;
  }
  const int dl = lane & 15;
  for (int q = 0; q < T_; ++q) {
    long long qr = ((long long)bn * T_ + q) * (3 * H_) + h * DH_;
    float s = 0.f;
#pragma unroll
    for (int d = 0; d < DH_; ++d) s += QKV[qr + d] * kv[d];
    s = s * 0.25f + kb;                       // 1/sqrt(16)
    float mx = wave_max(s);
    float p = __expf(s - mx);                 // masked / out-of-range lanes -> 0
    float sum = wave_sum(p);
    float a = p / sum;
    float accv = 0.f;
    for (int t2 = 0; t2 < T_; ++t2) {
      float at = __shfl(a, t2, 32);
      accv += at * QKV[((long long)bn * T_ + t2) * (3 * H_) + 2 * H_ + h * DH_ + dl];
    }
    if (lane < 16) o[((long long)bn * T_ + q) * H_ + h * DH_ + dl] = (bf16)accv;
  }
}

// x[bn][t][h] = H6[t][bn][h]*m[t][bn] + pos[t][h]
__global__ void assemble_kernel(const bf16* __restrict__ Hb, const float* __restrict__ maskf,
                                const float* __restrict__ pos, float* __restrict__ xF,
                                bf16* __restrict__ xB) {
  long long idx = (long long)blockIdx.x * 256 + threadIdx.x;
  if (idx >= (long long)BN_ * T_ * H_) return;
  int hh = (int)(idx % H_);
  long long r = idx / H_;
  int t = (int)(r % T_);
  long long bn = r / T_;
  float v = (float)Hb[((long long)t * BN_ + bn) * H_ + hh] * maskf[t * BN_ + bn] + pos[t * H_ + hh];
  xF[idx] = v;
  xB[idx] = (bf16)v;
}

// ---------------------------------------------------------------- launcher
extern "C" void kernel_launch(void* const* d_in, const int* in_sizes, int n_in,
                              void* d_out, int out_size, void* d_ws, size_t ws_size,
                              hipStream_t stream) {
  (void)in_sizes; (void)n_in; (void)out_size; (void)ws_size;
  const unsigned char* ego  = (const unsigned char*)d_in[0];
  const float* x_raw  = (const float*)d_in[1];
  const float* adj    = (const float*)d_in[2];
  const float* gcn1_w = (const float*)d_in[3];
  const float* gcn_ws = (const float*)d_in[4];
  const float* gcn_bs = (const float*)d_in[5];
  const float* ln_g   = (const float*)d_in[6];
  const float* ln_b   = (const float*)d_in[7];
  const float* pos    = (const float*)d_in[8];
  const float* t_wqkv = (const float*)d_in[9];
  const float* t_bqkv = (const float*)d_in[10];
  const float* t_wo   = (const float*)d_in[11];
  const float* t_bo   = (const float*)d_in[12];
  const float* t_ln1g = (const float*)d_in[13];
  const float* t_ln1b = (const float*)d_in[14];
  const float* t_w1   = (const float*)d_in[15];
  const float* t_b1   = (const float*)d_in[16];
  const float* t_w2   = (const float*)d_in[17];
  const float* t_b2   = (const float*)d_in[18];
  const float* t_ln2g = (const float*)d_in[19];
  const float* t_ln2b = (const float*)d_in[20];

  char* ws = (char*)d_ws;
  size_t off = 0;
  auto alloc = [&](size_t bytes) -> void* {
    void* p = ws + off;
    off = (off + bytes + 255) & ~(size_t)255;
    return p;
  };
  float* maskf = (float*)alloc(sizeof(float) * T_ * BN_);
  float* dinv  = (float*)alloc(sizeof(float) * T_ * BN_);
  bf16*  S     = (bf16*) alloc(sizeof(bf16) * (size_t)T_ * BN_ * BN_);   // 168 MB, L2-resident
  bf16*  Pt    = (bf16*) alloc(sizeof(bf16) * (size_t)ROWS_ * H_);       // [t][h][node]
  float* Z     = (float*)alloc(sizeof(float) * (size_t)ROWS_ * H_);
  bf16*  Hb    = (bf16*) alloc(sizeof(bf16) * (size_t)ROWS_ * H_);
  float* xF    = (float*)alloc(sizeof(float) * (size_t)ROWS_ * H_);
  bf16*  xB    = (bf16*) alloc(sizeof(bf16) * (size_t)ROWS_ * H_);
  float* QKVb  = (float*)alloc(sizeof(float) * (size_t)ROWS_ * 3 * H_);
  bf16*  oB    = (bf16*) alloc(sizeof(bf16) * (size_t)ROWS_ * H_);
  float* G     = (float*)alloc(sizeof(float) * (size_t)ROWS_ * H_);
  bf16*  H1    = (bf16*) alloc(sizeof(bf16) * (size_t)ROWS_ * FF_);
  bf16* gcnWt_b= (bf16*)alloc(sizeof(bf16) * 5L * H_ * H_);   // transposed (Wt[n][k])
  bf16* wqkv_b = (bf16*)alloc(sizeof(bf16) * 5L * 3 * H_ * H_);
  bf16* wo_b   = (bf16*)alloc(sizeof(bf16) * 5L * H_ * H_);
  bf16* w1_b   = (bf16*)alloc(sizeof(bf16) * 5L * FF_ * H_);
  bf16* w2_b   = (bf16*)alloc(sizeof(bf16) * 5L * H_ * FF_);

  auto cvt = [&](const float* src, bf16* dst, long long n) {
    f2bf_kernel<<<(int)((n + 255) / 256), 256, 0, stream>>>(src, dst, n);
  };
  f2bf_t_kernel<<<dim3((H_ * H_ + 255) / 256, 5), 256, 0, stream>>>(gcn_ws, gcnWt_b);
  cvt(t_wqkv, wqkv_b, 5LL * 3 * H_ * H_);
  cvt(t_wo,   wo_b,   5LL * H_ * H_);
  cvt(t_w1,   w1_b,   5LL * FF_ * H_);
  cvt(t_w2,   w2_b,   5LL * H_ * FF_);

  // ---------- GCN stage ----------
  mask_kernel<<<(T_ * BN_ + 255) / 256, 256, 0, stream>>>(ego, maskf);
  deg_dinv_kernel<<<dim3(BN_ / 256, T_), 256, 0, stream>>>(adj, maskf, dinv);
  build_S_kernel<<<dim3(BN_ / 64, BN_ / 64, T_), 256, 0, stream>>>(adj, dinv, S);
  gcn_in_kernel<<<(int)(((long long)ROWS_ * H_ + 255) / 256), 256, 0, stream>>>(x_raw, gcn1_w, Pt);

  auto Sgemm = [&]() {  // Z[t] = S[t] @ Pt[t]^T
    wmma_gemm_kernel<false, false, false>
        <<<dim3(BN_ / 64, H_ / 64, T_), 256, 0, stream>>>(
            S, Pt, nullptr, (void*)Z, BN_, H_, BN_, BN_, BN_, H_,
            (long long)BN_ * BN_, (long long)H_ * BN_, (long long)BN_ * H_);
  };

  // layer 0: h = relu(LN(S@(x@W1) + b0))
  Sgemm();
  ln_kernel<<<ROWS_ / 8, 256, 0, stream>>>(Z, gcn_bs, nullptr, nullptr,
                                           ln_g, ln_b, nullptr, Hb, ROWS_, 1);
  // layers 1..5: h = relu(LN(S@(h@Wi) + b_{i+1}) + h)
  for (int i = 0; i < 5; ++i) {
    // Pt = (Hb @ Wi) stored transposed per-t
    wmma_gemm_kernel<false, true, true>
        <<<dim3(ROWS_ / 64, H_ / 64, 1), 256, 0, stream>>>(
            Hb, gcnWt_b + (long long)i * H_ * H_, nullptr, (void*)Pt,
            ROWS_, H_, H_, H_, H_, H_, 0, 0, 0);
    Sgemm();
    ln_kernel<<<ROWS_ / 8, 256, 0, stream>>>(
        Z, gcn_bs + (i + 1) * H_, nullptr, Hb,
        ln_g + (i + 1) * H_, ln_b + (i + 1) * H_, nullptr, Hb, ROWS_, 1);
  }
  assemble_kernel<<<(int)(((long long)ROWS_ * H_ + 255) / 256), 256, 0, stream>>>(
      Hb, maskf, pos, xF, xB);

  // ---------- Transformer stage ----------
  for (int l = 0; l < LAYERS_; ++l) {
    // QKV = x @ Wqkv^T + bqkv
    wmma_gemm_kernel<false, false, false>
        <<<dim3(ROWS_ / 64, (3 * H_) / 64, 1), 256, 0, stream>>>(
            xB, wqkv_b + (long long)l * 3 * H_ * H_, t_bqkv + l * 3 * H_,
            (void*)QKVb, ROWS_, 3 * H_, H_, H_, H_, 3 * H_, 0, 0, 0);
    attn_kernel<<<BN_ * HEADS_, 32, 0, stream>>>(QKVb, maskf, oB);
    // O = o @ Wo^T  (bias folded into LN1)
    wmma_gemm_kernel<false, false, false>
        <<<dim3(ROWS_ / 64, H_ / 64, 1), 256, 0, stream>>>(
            oB, wo_b + (long long)l * H_ * H_, nullptr,
            (void*)G, ROWS_, H_, H_, H_, H_, H_, 0, 0, 0);
    ln_kernel<<<ROWS_ / 8, 256, 0, stream>>>(
        G, t_bo + l * H_, xF, nullptr, t_ln1g + l * H_, t_ln1b + l * H_,
        xF, xB, ROWS_, 0);
    // H1 = relu(x @ W1^T + b1)
    wmma_gemm_kernel<true, true, false>
        <<<dim3(ROWS_ / 64, FF_ / 64, 1), 256, 0, stream>>>(
            xB, w1_b + (long long)l * FF_ * H_, t_b1 + l * FF_,
            (void*)H1, ROWS_, FF_, H_, H_, H_, FF_, 0, 0, 0);
    // F2 = H1 @ W2^T (bias folded into LN2)
    wmma_gemm_kernel<false, false, false>
        <<<dim3(ROWS_ / 64, H_ / 64, 1), 256, 0, stream>>>(
            H1, w2_b + (long long)l * H_ * FF_, nullptr,
            (void*)G, ROWS_, H_, FF_, FF_, FF_, H_, 0, 0, 0);
    const bool last = (l == LAYERS_ - 1);
    ln_kernel<<<ROWS_ / 8, 256, 0, stream>>>(
        G, t_b2 + l * H_, xF, nullptr, t_ln2g + l * H_, t_ln2b + l * H_,
        last ? (float*)d_out : xF, last ? nullptr : xB, ROWS_, 0);
  }
}
